// AutoNavModel_11398843204125
// MI455X (gfx1250) — matrix operand order repaired and checked by reading, not compiled, not run on hardware
//
#include <hip/hip_runtime.h>

#define C_DIM 256
#define ND 900
#define NM 100
#define NPTS 20

typedef __bf16 bf16_t;
typedef __bf16 v16bf __attribute__((ext_vector_type(16)));
typedef __bf16 v8bf  __attribute__((ext_vector_type(8)));
typedef float  v8f   __attribute__((ext_vector_type(8)));

// ---------------------------------------------------------------------------
// Stage 1: projection + multiscale bilinear sampling -> agg rows in bf16.
// Rows 0..899 = det anchors, 900..911 = zero pad, 912..1011 = map centers,
// 1012..1023 = zero pad. One block per row, one thread per channel.
// ---------------------------------------------------------------------------
__device__ __forceinline__ float bilin_sample(const float* __restrict__ base,
                                              int W, int H, float gx, float gy) {
  float xs = (gx + 1.f) * (W * 0.5f) - 0.5f;
  float ys = (gy + 1.f) * (H * 0.5f) - 0.5f;
  xs = fminf(fmaxf(xs, -1.0e8f), 1.0e8f);
  ys = fminf(fmaxf(ys, -1.0e8f), 1.0e8f);
  float x0f = floorf(xs), y0f = floorf(ys);
  float wx1 = xs - x0f, wx0 = 1.f - wx1;
  float wy1 = ys - y0f, wy0 = 1.f - wy1;
  int x0 = (int)x0f, y0 = (int)y0f;
  int x1 = x0 + 1,  y1 = y0 + 1;
  float s = 0.f;
  if (x0 >= 0 && x0 < W && y0 >= 0 && y0 < H) s += base[y0 * W + x0] * (wx0 * wy0);
  if (x1 >= 0 && x1 < W && y0 >= 0 && y0 < H) s += base[y0 * W + x1] * (wx1 * wy0);
  if (x0 >= 0 && x0 < W && y1 >= 0 && y1 < H) s += base[y1 * W + x0] * (wx0 * wy1);
  if (x1 >= 0 && x1 < W && y1 >= 0 && y1 < H) s += base[y1 * W + x1] * (wx1 * wy1);
  return s;
}

__global__ __launch_bounds__(256)
void sample_agg_kernel(const float* __restrict__ p2, const float* __restrict__ p3,
                       const float* __restrict__ p4, const float* __restrict__ p5,
                       const float* __restrict__ K6, const float* __restrict__ E6,
                       const float* __restrict__ det3d, const float* __restrict__ mapA,
                       bf16_t* __restrict__ agg) {
  int b  = blockIdx.x;   // 0..1023
  int ch = threadIdx.x;  // 0..255
  bool pad = (b >= ND && b < 912) || (b >= 912 + NM);
  if (pad) { agg[(size_t)b * C_DIM + ch] = (bf16_t)0.f; return; }

  float hx, hy, hz;
  if (b < ND) {
    hx = det3d[b * 3 + 0]; hy = det3d[b * 3 + 1]; hz = det3d[b * 3 + 2];
  } else {
    int m = b - 912;
    float cx = 0.f, cy = 0.f;
    for (int p = 0; p < NPTS; ++p) {
      cx += mapA[(m * NPTS + p) * 2 + 0];
      cy += mapA[(m * NPTS + p) * 2 + 1];
    }
    hx = cx * (1.f / NPTS); hy = cy * (1.f / NPTS); hz = 0.f;
  }

  const float* feats[4] = {p2, p3, p4, p5};
  const int Ws[4] = {160, 80, 40, 20};
  const int Hs[4] = {120, 60, 30, 15};

  float acc = 0.f;
  for (int c = 0; c < 6; ++c) {
    const float* E  = E6 + c * 16;
    const float* Kc = K6 + c * 9;
    float pc0 = E[0] * hx + E[1] * hy + E[2]  * hz + E[3];
    float pc1 = E[4] * hx + E[5] * hy + E[6]  * hz + E[7];
    float pc2 = E[8] * hx + E[9] * hy + E[10] * hz + E[11];
    float q0 = Kc[0] * pc0 + Kc[1] * pc1 + Kc[2] * pc2;
    float q1 = Kc[3] * pc0 + Kc[4] * pc1 + Kc[5] * pc2;
    float q2 = Kc[6] * pc0 + Kc[7] * pc1 + Kc[8] * pc2;
    if (!(q2 > 0.1f)) continue;  // invalid camera contributes zero
    float inv = 1.f / (q2 + 1e-6f);
    float gx = (q0 * inv) * (2.f / 640.f) - 1.f;
    float gy = (q1 * inv) * (2.f / 480.f) - 1.f;
    float lsum = 0.f;
    for (int l = 0; l < 4; ++l) {
      int W = Ws[l], H = Hs[l];
      const float* base = feats[l] + ((size_t)c * C_DIM + ch) * (size_t)(W * H);
      lsum += bilin_sample(base, W, H, gx, gy);
    }
    acc += lsum * 0.25f;
  }
  acc *= (1.f / 6.f);
  agg[(size_t)b * C_DIM + ch] = (bf16_t)acc;
}

// ---------------------------------------------------------------------------
// Stage 2a: weight transpose+convert: W[k][n] f32 -> Wt[n][k] bf16, so each
// WMMA B fragment lane reads one contiguous 32B chunk.
// ---------------------------------------------------------------------------
__global__ __launch_bounds__(256)
void wt_bf16_kernel(const float* __restrict__ Wd, const float* __restrict__ Wm,
                    bf16_t* __restrict__ WtD, bf16_t* __restrict__ WtM) {
  int n = blockIdx.x & 255;
  int t = threadIdx.x;
  if (blockIdx.x < 256) WtD[(size_t)n * C_DIM + t] = (bf16_t)Wd[(size_t)t * C_DIM + n];
  else                  WtM[(size_t)n * C_DIM + t] = (bf16_t)Wm[(size_t)t * C_DIM + n];
}

// ---------------------------------------------------------------------------
// Stage 2b: H = relu(A @ W + bias), A bf16 [1024][256], Wt bf16 [N][K],
// one wave per 16x16 tile, v_wmma_f32_16x16x32_bf16, K unrolled 8x.
// ---------------------------------------------------------------------------
__global__ __launch_bounds__(256)
void wmma_gemm_relu(const bf16_t* __restrict__ A, const bf16_t* __restrict__ Wt,
                    const float* __restrict__ bias, float* __restrict__ Hout,
                    int rowBase, int mTiles) {
  int wave = threadIdx.x >> 5;
  int lane = threadIdx.x & 31;
  int tile = blockIdx.x * 8 + wave;
  if (tile >= mTiles * 16) return;          // wave-uniform guard
  int mt = tile >> 4, nt = tile & 15;
  int hi  = lane >> 4;                      // which half of the wave
  int l15 = lane & 15;
  int row = rowBase + mt * 16 + l15;        // A row held by this lane
  int col = nt * 16 + l15;                  // B column held by this lane
  const bf16_t* arow = A  + (size_t)row * C_DIM;
  const bf16_t* brow = Wt + (size_t)col * C_DIM;

  v8f acc = {};
#pragma unroll
  for (int k0 = 0; k0 < C_DIM; k0 += 32) {
    // A fragment: K = k0+hi*16 .. +15 of row `row` (contiguous, matches B)
    v16bf a = *(const v16bf*)(arow + k0 + hi * 16);
    // B fragment: K = k0+hi*16 .. +15 of column `col` (contiguous in Wt)
    v16bf b = *(const v16bf*)(brow + k0 + hi * 16);
    acc = __builtin_amdgcn_wmma_f32_16x16x32_bf16(
        false, a, false, b, (short)0, acc, false, false);
  }

  float bn = bias[col];
#pragma unroll
  for (int r = 0; r < 8; ++r) {             // D vgpr r <-> M = r + 8*hi
    int m = rowBase + mt * 16 + r + hi * 8;
    float val = fmaxf(acc[r] + bn, 0.f);
    Hout[(size_t)m * C_DIM + col] = val;
  }
}

// ---------------------------------------------------------------------------
// Stage 3: tiny linear heads + anchor residuals. One block per anchor.
// Output layout: det_classes[900*4] | det_boxes[900*11] | map_classes[100*3]
//                | map_lines[100*40]
// ---------------------------------------------------------------------------
__global__ __launch_bounds__(256)
void heads_kernel(const float* __restrict__ H,
                  const float* __restrict__ Wcd, const float* __restrict__ bcd,
                  const float* __restrict__ Wod, const float* __restrict__ bod,
                  const float* __restrict__ Wcm, const float* __restrict__ bcm,
                  const float* __restrict__ Wom, const float* __restrict__ bom,
                  const float* __restrict__ detFull, const float* __restrict__ mapA,
                  float* __restrict__ out) {
  __shared__ float sh[C_DIM];
  int b = blockIdx.x;
  int t = threadIdx.x;
  int hrow = (b < ND) ? b : (912 + (b - ND));
  sh[t] = H[(size_t)hrow * C_DIM + t];
  __syncthreads();

  if (b < ND) {
    if (t < 15) {
      float s = 0.f;
      if (t < 4) {
        for (int k = 0; k < C_DIM; ++k) s += sh[k] * Wcd[k * 4 + t];
        out[b * 4 + t] = s + bcd[t];
      } else {
        int o = t - 4;
        for (int k = 0; k < C_DIM; ++k) s += sh[k] * Wod[k * 11 + o];
        out[3600 + b * 11 + o] = detFull[b * 11 + o] + s + bod[o];
      }
    }
  } else {
    int m = b - ND;
    if (t < 43) {
      float s = 0.f;
      if (t < 3) {
        for (int k = 0; k < C_DIM; ++k) s += sh[k] * Wcm[k * 3 + t];
        out[13500 + m * 3 + t] = s + bcm[t];
      } else {
        int o = t - 3;
        for (int k = 0; k < C_DIM; ++k) s += sh[k] * Wom[k * 40 + o];
        out[13800 + m * 40 + o] = mapA[m * 40 + o] + s + bom[o];
      }
    }
  }
}

// ---------------------------------------------------------------------------
extern "C" void kernel_launch(void* const* d_in, const int* in_sizes, int n_in,
                              void* d_out, int out_size, void* d_ws, size_t ws_size,
                              hipStream_t stream) {
  const float* p2     = (const float*)d_in[0];
  const float* p3     = (const float*)d_in[1];
  const float* p4     = (const float*)d_in[2];
  const float* p5     = (const float*)d_in[3];
  const float* intr   = (const float*)d_in[4];
  const float* extr   = (const float*)d_in[5];
  const float* det3d  = (const float*)d_in[6];
  const float* detFul = (const float*)d_in[7];
  const float* mapA   = (const float*)d_in[8];
  const float* W1d    = (const float*)d_in[9];
  const float* b1d    = (const float*)d_in[10];
  const float* Wcd    = (const float*)d_in[11];
  const float* bcd    = (const float*)d_in[12];
  const float* Wod    = (const float*)d_in[13];
  const float* bod    = (const float*)d_in[14];
  const float* W1m    = (const float*)d_in[15];
  const float* b1m    = (const float*)d_in[16];
  const float* Wcm    = (const float*)d_in[17];
  const float* bcm    = (const float*)d_in[18];
  const float* Wom    = (const float*)d_in[19];
  const float* bom    = (const float*)d_in[20];

  char* ws = (char*)d_ws;
  bf16_t* agg = (bf16_t*)(ws);                 // 1024*256*2 = 524288 B
  bf16_t* WtD = (bf16_t*)(ws + 524288);        // 256*256*2  = 131072 B
  bf16_t* WtM = (bf16_t*)(ws + 655360);        // 256*256*2  = 131072 B
  float*  Hbuf = (float*)(ws + 786432);        // 1024*256*4 = 1048576 B

  sample_agg_kernel<<<1024, 256, 0, stream>>>(p2, p3, p4, p5, intr, extr,
                                              det3d, mapA, agg);
  wt_bf16_kernel<<<512, 256, 0, stream>>>(W1d, W1m, WtD, WtM);
  // det: rows 0..911 -> 57 M-tiles * 16 N-tiles = 912 tiles = 114 blocks
  wmma_gemm_relu<<<114, 256, 0, stream>>>(agg, WtD, b1d, Hbuf, 0, 57);
  // map: rows 912..1023 -> 7 * 16 = 112 tiles = 14 blocks
  wmma_gemm_relu<<<14, 256, 0, stream>>>(agg, WtM, b1m, Hbuf, 912, 7);
  heads_kernel<<<1000, 256, 0, stream>>>(Hbuf, Wcd, bcd, Wod, bod,
                                         Wcm, bcm, Wom, bom,
                                         detFul, mapA, (float*)d_out);
}
